// MultiHeadSelfAttention_36988258353748
// MI455X (gfx1250) — matrix operand order repaired and checked by reading, not compiled, not run on hardware
//
#include <hip/hip_runtime.h>

typedef _Float16 half_t;
typedef __attribute__((ext_vector_type(16))) _Float16 v16h;
typedef __attribute__((ext_vector_type(8)))  _Float16 v8h;
typedef __attribute__((ext_vector_type(8)))  float    v8f;

#define HDIM   768
#define NHEADS 12
#define HEADD  64
#define SEQ    2048
#define BATCH  2

__device__ __forceinline__ v16h pack16(v8h lo, v8h hi) {
  v16h r;
#pragma unroll
  for (int i = 0; i < 8; ++i) { r[i] = lo[i]; r[8 + i] = hi[i]; }
  return r;
}

__device__ __forceinline__ v8f zero8() {
  v8f z;
#pragma unroll
  for (int i = 0; i < 8; ++i) z[i] = 0.0f;
  return z;
}

__device__ __forceinline__ v8f wmma_f16(v16h a, v16h b, v8f c) {
  // D = A(16x32 f16) * B(32x16 f16) + C(16x16 f32)
  return __builtin_amdgcn_wmma_f32_16x16x32_f16(false, a, false, b, (short)0, c,
                                                false, false);
}

// Flat pointers to LDS are {shared-aperture-hi32, lds-byte-offset-lo32}; the
// async-to-LDS VDST operand wants the lds byte offset, i.e. the low 32 bits.
__device__ __forceinline__ unsigned lds_off32(const void* p) {
  return (unsigned)(unsigned long long)p;
}

// global -> LDS direct copy, 16 bytes per lane, tracked by ASYNCcnt.
__device__ __forceinline__ void async_copy_b128(unsigned lds_byte_off,
                                                const void* gaddr) {
  asm volatile("global_load_async_to_lds_b128 %0, %1, off"
               :: "v"(lds_byte_off), "v"(gaddr) : "memory");
}

// ---------------------------------------------------------------------------
// Kernel 1: fused QKV projection.  C[4096, 2304] = X[4096,768] @ [Wq|Wk|Wv]
// Block tile 64(M) x 256(N), 8 waves as 2x4, wave tile 32x64, K-step 32.
// Q,K stored f16 [b,h,s,d]; V stored f16 transposed [b,h,d,s].
// ---------------------------------------------------------------------------
__global__ __launch_bounds__(256) void qkv_proj_kernel(
    const float* __restrict__ X,
    const float* __restrict__ Wq, const float* __restrict__ bq,
    const float* __restrict__ Wk, const float* __restrict__ bk,
    const float* __restrict__ Wv, const float* __restrict__ bv,
    half_t* __restrict__ qws, half_t* __restrict__ kws, half_t* __restrict__ vws)
{
  constexpr int ASTR = 40;  // halves per A-LDS row (padded from 32)
  constexpr int BSTR = 40;  // halves per B^T-LDS row (padded from 32)
  __shared__ half_t As[64 * ASTR];    // [m][k]
  __shared__ half_t Bs[256 * BSTR];   // [n][k]  (transposed weight tile)

  const int t = threadIdx.x;
  const int wave = t >> 5, lane = t & 31;
  const int wm = wave >> 2, wn = wave & 3;
  const int col = lane & 15, lh = lane >> 4;

  const int blockM = blockIdx.x * 64;
  const int nTile  = blockIdx.y;            // 0..8 ; 256-wide tiles over 2304
  const int which  = nTile / 3;             // 0=Q 1=K 2=V (768/256==3, no split)
  const int blockN = (nTile % 3) * 256;     // column offset inside the matrix
  const float* __restrict__ W    = (which == 0) ? Wq : (which == 1) ? Wk : Wv;
  const float* __restrict__ bias = (which == 0) ? bq : (which == 1) ? bk : bv;

  v8f acc[2][4];
#pragma unroll
  for (int i = 0; i < 2; ++i)
#pragma unroll
    for (int j = 0; j < 4; ++j) acc[i][j] = zero8();

  const int a_row = t >> 2, a_kp = (t & 3) * 8;        // 64 rows x 32 k
  const int b_n0  = (t & 63) * 4, b_k0 = (t >> 6) * 8; // 256 n x 32 k

  for (int kb = 0; kb < HDIM; kb += 32) {
    __syncthreads();
    // --- stage A: X f32 -> f16 LDS [m][k]
    {
      const float4* src = (const float4*)(X + (size_t)(blockM + a_row) * HDIM + kb + a_kp);
      float4 f0 = src[0], f1 = src[1];
      v8h h;
      h[0] = (half_t)f0.x; h[1] = (half_t)f0.y; h[2] = (half_t)f0.z; h[3] = (half_t)f0.w;
      h[4] = (half_t)f1.x; h[5] = (half_t)f1.y; h[6] = (half_t)f1.z; h[7] = (half_t)f1.w;
      *(v8h*)&As[a_row * ASTR + a_kp] = h;
    }
    // --- stage B transposed: W f32 [k][n] -> f16 LDS [n][k]
    {
      float4 colv[8];
#pragma unroll
      for (int kk = 0; kk < 8; ++kk)
        colv[kk] = *(const float4*)(W + (size_t)(kb + b_k0 + kk) * HDIM + blockN + b_n0);
#pragma unroll
      for (int i = 0; i < 4; ++i) {
        v8h h;
#pragma unroll
        for (int kk = 0; kk < 8; ++kk) {
          const float* cp = reinterpret_cast<const float*>(&colv[kk]);
          h[kk] = (half_t)cp[i];
        }
        *(v8h*)&Bs[(b_n0 + i) * BSTR + b_k0] = h;
      }
    }
    // warm L2/WGP$ for the next K-step while this one computes
    if (kb + 32 < HDIM) {
      __builtin_prefetch(X + (size_t)(blockM + a_row) * HDIM + kb + 32 + a_kp, 0, 0);
      __builtin_prefetch(W + (size_t)(kb + 32 + b_k0) * HDIM + blockN + b_n0, 0, 0);
    }
    __syncthreads();

    v16h afr[2], bfr[4];
#pragma unroll
    for (int mf = 0; mf < 2; ++mf) {
      const int ar = wm * 32 + mf * 16 + col;
      v8h lo = *(const v8h*)&As[ar * ASTR + lh * 8];
      v8h hi = *(const v8h*)&As[ar * ASTR + 16 + lh * 8];
      afr[mf] = pack16(lo, hi);
    }
#pragma unroll
    for (int nf = 0; nf < 4; ++nf) {
      const int bn = wn * 64 + nf * 16 + col;
      v8h lo = *(const v8h*)&Bs[bn * BSTR + lh * 8];
      v8h hi = *(const v8h*)&Bs[bn * BSTR + 16 + lh * 8];
      bfr[nf] = pack16(lo, hi);
    }
#pragma unroll
    for (int mf = 0; mf < 2; ++mf)
#pragma unroll
      for (int nf = 0; nf < 4; ++nf)
        acc[mf][nf] = wmma_f16(afr[mf], bfr[nf], acc[mf][nf]);
  }

  // --- epilogue: scatter into head-major layouts
#pragma unroll
  for (int mf = 0; mf < 2; ++mf) {
#pragma unroll
    for (int nf = 0; nf < 4; ++nf) {
      const int n_in = blockN + wn * 64 + nf * 16 + col;   // 0..767
      const int hh = n_in >> 6, dd = n_in & 63;
      const float badd = bias[n_in];
#pragma unroll
      for (int r = 0; r < 8; ++r) {
        const int mg = blockM + wm * 32 + mf * 16 + lh * 8 + r;
        const int bb = mg >> 11, ss = mg & 2047;
        const float val = acc[mf][nf][r] + badd;
        if (which == 0)
          qws[((size_t)(bb * NHEADS + hh) * SEQ + ss) * HEADD + dd] = (half_t)val;
        else if (which == 1)
          kws[((size_t)(bb * NHEADS + hh) * SEQ + ss) * HEADD + dd] = (half_t)val;
        else
          vws[((size_t)(bb * NHEADS + hh) * HEADD + dd) * SEQ + ss] = (half_t)val;
      }
    }
  }
}

// ---------------------------------------------------------------------------
// Kernel 2: flash attention per (q-tile, head, batch). 128 threads = 4 waves,
// each wave owns 16 q rows; 64-key tiles double-buffered in LDS via the
// CDNA5 async global->LDS path (ASYNCcnt), overlapping HBM with WMMA.
// ---------------------------------------------------------------------------
constexpr int ATTN_STR = 72;   // padded LDS row stride in halves (144B)

__device__ __forceinline__ void stage_tiles_async(
    const half_t* __restrict__ Kb, const half_t* __restrict__ Vb, int kt,
    half_t* KsBuf, half_t* VsBuf, int srow, int spart)
{
  // 128 threads: thread handles 64B of one K row and 64B of one V row
  const half_t* gk = Kb + (size_t)(kt + srow) * HEADD + spart;   // [key][d]
  const half_t* gv = Vb + (size_t)srow * SEQ + kt + spart;       // [d][s]
  const unsigned lk = lds_off32(KsBuf + srow * ATTN_STR + spart);
  const unsigned lv = lds_off32(VsBuf + srow * ATTN_STR + spart);
#pragma unroll
  for (int i = 0; i < 4; ++i) {
    async_copy_b128(lk + i * 16, gk + i * 8);
    async_copy_b128(lv + i * 16, gv + i * 8);
  }
}

__global__ __launch_bounds__(128) void attn_kernel(
    const half_t* __restrict__ qws, const half_t* __restrict__ kws,
    const half_t* __restrict__ vws, const float* __restrict__ mask,
    half_t* __restrict__ ctx)
{
  constexpr int STR = ATTN_STR;
  __shared__ half_t Ks[2][64 * STR];      // [key][d], double buffered
  __shared__ half_t Vs[2][64 * STR];      // [d][key], double buffered
  __shared__ half_t Ps[4 * 16 * STR];     // per-wave P tile [qrow][key]

  const int t = threadIdx.x;
  const int wave = t >> 5, lane = t & 31;
  const int col = lane & 15, lh = lane >> 4;
  const int qt = blockIdx.x, hh = blockIdx.y, bb = blockIdx.z;
  const int bh = bb * NHEADS + hh;

  const half_t* __restrict__ Qb = qws + (size_t)bh * SEQ * HEADD;
  const half_t* __restrict__ Kb = kws + (size_t)bh * SEQ * HEADD;
  const half_t* __restrict__ Vb = vws + (size_t)bh * HEADD * SEQ;
  const float*  __restrict__ mrow = mask + (size_t)bb * SEQ;

  const int qbase = qt * 64 + wave * 16;

  // Q fragments (persistent): A-matrix 16x32, two K-steps cover HD=64
  v16h qf[2];
#pragma unroll
  for (int ks = 0; ks < 2; ++ks) {
    const int koff = ks * 32;
    const half_t* qp = Qb + (size_t)(qbase + col) * HEADD;
    v8h lo = *(const v8h*)(qp + koff + lh * 8);
    v8h hi = *(const v8h*)(qp + koff + 16 + lh * 8);
    qf[ks] = pack16(lo, hi);
  }

  v8f of[4];
#pragma unroll
  for (int i = 0; i < 4; ++i) of[i] = zero8();
  float m_r[8], l_r[8];
#pragma unroll
  for (int r = 0; r < 8; ++r) { m_r[r] = -1e30f; l_r[r] = 0.0f; }

  const int srow = t >> 1, spart = (t & 1) * 32;  // 64 rows x 64 halves / 128 thr

  // prologue: launch tile 0 (8 async ops per wave)
  stage_tiles_async(Kb, Vb, 0, Ks[0], Vs[0], srow, spart);

  for (int kt = 0, it = 0; kt < SEQ; kt += 64, ++it) {
    const int cur = it & 1;
    __syncthreads();   // all waves done reading the buffer we are about to refill
    if (kt + 64 < SEQ) {
      stage_tiles_async(Kb, Vb, kt + 64, Ks[cur ^ 1], Vs[cur ^ 1], srow, spart);
      // 8 new ops in flight; drain down to them => current tile complete
      asm volatile("s_wait_asynccnt 0x8" ::: "memory");
    } else {
      asm volatile("s_wait_asynccnt 0x0" ::: "memory");
    }
    __syncthreads();   // current tile visible to every wave

    const half_t* Kc = Ks[cur];
    const half_t* Vc = Vs[cur];

    // scores: S = Q @ K^T ; B-frag element e <-> d=koff+lh*8+e (e<8), koff+16+lh*8+(e-8) (e>=8), col <-> key
    float sv[4][8];
#pragma unroll
    for (int kf = 0; kf < 4; ++kf) {
      v8f acc = zero8();
#pragma unroll
      for (int ks = 0; ks < 2; ++ks) {
        const int koff = ks * 32;
        const int key = kf * 16 + col;
        v8h lo = *(const v8h*)&Kc[key * STR + koff + lh * 8];
        v8h hi = *(const v8h*)&Kc[key * STR + koff + 16 + lh * 8];
        acc = wmma_f16(qf[ks], pack16(lo, hi), acc);
      }
      const float mv = mrow[kt + kf * 16 + col];
#pragma unroll
      for (int r = 0; r < 8; ++r) sv[kf][r] = acc[r] * 0.125f + mv;
    }

    // online softmax (rows live in lanes {0-15} / {16-31} separately)
    float tmax[8];
#pragma unroll
    for (int r = 0; r < 8; ++r)
      tmax[r] = fmaxf(fmaxf(sv[0][r], sv[1][r]), fmaxf(sv[2][r], sv[3][r]));
#pragma unroll
    for (int off = 1; off < 16; off <<= 1)
#pragma unroll
      for (int r = 0; r < 8; ++r)
        tmax[r] = fmaxf(tmax[r], __shfl_xor(tmax[r], off, 32));

    float mnew[8], alpha[8], ps[8];
#pragma unroll
    for (int r = 0; r < 8; ++r) {
      mnew[r]  = fmaxf(m_r[r], tmax[r]);
      alpha[r] = __expf(m_r[r] - mnew[r]);
      ps[r]    = 0.0f;
    }
#pragma unroll
    for (int kf = 0; kf < 4; ++kf) {
      const int key = kf * 16 + col;
#pragma unroll
      for (int r = 0; r < 8; ++r) {
        const float p = __expf(sv[kf][r] - mnew[r]);
        ps[r] += p;
        Ps[(wave * 16 + lh * 8 + r) * STR + key] = (half_t)p;  // C-layout -> LDS
      }
    }
#pragma unroll
    for (int off = 1; off < 16; off <<= 1)
#pragma unroll
      for (int r = 0; r < 8; ++r)
        ps[r] += __shfl_xor(ps[r], off, 32);
#pragma unroll
    for (int r = 0; r < 8; ++r) {
      l_r[r] = l_r[r] * alpha[r] + ps[r];
      m_r[r] = mnew[r];
    }
#pragma unroll
    for (int df = 0; df < 4; ++df)
#pragma unroll
      for (int r = 0; r < 8; ++r) of[df][r] *= alpha[r];

    // O += P @ V ; re-read P in A-layout (LDS in-order within a wave)
#pragma unroll
    for (int ks = 0; ks < 2; ++ks) {
      const int koff = ks * 32;
      const half_t* prow = &Ps[(wave * 16 + col) * STR];
      v8h plo = *(const v8h*)(prow + koff + lh * 8);
      v8h phi = *(const v8h*)(prow + koff + 16 + lh * 8);
      v16h pf = pack16(plo, phi);
#pragma unroll
      for (int df = 0; df < 4; ++df) {
        const int dcol = df * 16 + col;
        v8h vlo = *(const v8h*)&Vc[dcol * STR + koff + lh * 8];
        v8h vhi = *(const v8h*)&Vc[dcol * STR + koff + 16 + lh * 8];
        of[df] = wmma_f16(pf, pack16(vlo, vhi), of[df]);
      }
    }
  }

  // epilogue: ctx[b][s][h*64+d] f16
#pragma unroll
  for (int df = 0; df < 4; ++df) {
    const int dcol = df * 16 + col;
#pragma unroll
    for (int r = 0; r < 8; ++r) {
      const int q = qbase + lh * 8 + r;
      const float v = of[df][r] / l_r[r];
      ctx[((size_t)bb * SEQ + q) * HDIM + hh * HEADD + dcol] = (half_t)v;
    }
  }
}

// ---------------------------------------------------------------------------
// Kernel 3: output projection. out[4096,768] = ctx(f16) @ Wo + bo  (f32 out)
// ---------------------------------------------------------------------------
__global__ __launch_bounds__(256) void out_proj_kernel(
    const half_t* __restrict__ Xh, const float* __restrict__ Wo,
    const float* __restrict__ bo, float* __restrict__ out)
{
  constexpr int ASTR = 40, BSTR = 40;
  __shared__ half_t As[64 * ASTR];
  __shared__ half_t Bs[256 * BSTR];

  const int t = threadIdx.x;
  const int wave = t >> 5, lane = t & 31;
  const int wm = wave >> 2, wn = wave & 3;
  const int col = lane & 15, lh = lane >> 4;

  const int blockM = blockIdx.x * 64;
  const int blockN = blockIdx.y * 256;

  v8f acc[2][4];
#pragma unroll
  for (int i = 0; i < 2; ++i)
#pragma unroll
    for (int j = 0; j < 4; ++j) acc[i][j] = zero8();

  const int a_row = t >> 2, a_kp = (t & 3) * 8;
  const int b_n0  = (t & 63) * 4, b_k0 = (t >> 6) * 8;

  for (int kb = 0; kb < HDIM; kb += 32) {
    __syncthreads();
    {
      v8h h = *(const v8h*)(Xh + (size_t)(blockM + a_row) * HDIM + kb + a_kp);
      *(v8h*)&As[a_row * ASTR + a_kp] = h;
    }
    {
      float4 colv[8];
#pragma unroll
      for (int kk = 0; kk < 8; ++kk)
        colv[kk] = *(const float4*)(Wo + (size_t)(kb + b_k0 + kk) * HDIM + blockN + b_n0);
#pragma unroll
      for (int i = 0; i < 4; ++i) {
        v8h h;
#pragma unroll
        for (int kk = 0; kk < 8; ++kk) {
          const float* cp = reinterpret_cast<const float*>(&colv[kk]);
          h[kk] = (half_t)cp[i];
        }
        *(v8h*)&Bs[(b_n0 + i) * BSTR + b_k0] = h;
      }
    }
    if (kb + 32 < HDIM) {
      __builtin_prefetch(Xh + (size_t)(blockM + a_row) * HDIM + kb + 32 + a_kp, 0, 0);
      __builtin_prefetch(Wo + (size_t)(kb + 32 + b_k0) * HDIM + blockN + b_n0, 0, 0);
    }
    __syncthreads();

    v16h afr[2], bfr[4];
#pragma unroll
    for (int mf = 0; mf < 2; ++mf) {
      const int ar = wm * 32 + mf * 16 + col;
      v8h lo = *(const v8h*)&As[ar * ASTR + lh * 8];
      v8h hi = *(const v8h*)&As[ar * ASTR + 16 + lh * 8];
      afr[mf] = pack16(lo, hi);
    }
#pragma unroll
    for (int nf = 0; nf < 4; ++nf) {
      const int bn = wn * 64 + nf * 16 + col;
      v8h lo = *(const v8h*)&Bs[bn * BSTR + lh * 8];
      v8h hi = *(const v8h*)&Bs[bn * BSTR + 16 + lh * 8];
      bfr[nf] = pack16(lo, hi);
    }
#pragma unroll
    for (int mf = 0; mf < 2; ++mf)
#pragma unroll
      for (int nf = 0; nf < 4; ++nf)
        acc[mf][nf] = wmma_f16(afr[mf], bfr[nf], acc[mf][nf]);
  }

#pragma unroll
  for (int mf = 0; mf < 2; ++mf) {
#pragma unroll
    for (int nf = 0; nf < 4; ++nf) {
      const int n = blockN + wn * 64 + nf * 16 + col;
      const float badd = bo[n];
#pragma unroll
      for (int r = 0; r < 8; ++r) {
        const int mg = blockM + wm * 32 + mf * 16 + lh * 8 + r;
        out[(size_t)mg * HDIM + n] = acc[mf][nf][r] + badd;
      }
    }
  }
}

// ---------------------------------------------------------------------------
extern "C" void kernel_launch(void* const* d_in, const int* in_sizes, int n_in,
                              void* d_out, int out_size, void* d_ws, size_t ws_size,
                              hipStream_t stream) {
  (void)in_sizes; (void)n_in; (void)out_size; (void)ws_size;
  const float* hs   = (const float*)d_in[0];
  const float* mask = (const float*)d_in[1];
  const float* Wq = (const float*)d_in[2];
  const float* bq = (const float*)d_in[3];
  const float* Wk = (const float*)d_in[4];
  const float* bk = (const float*)d_in[5];
  const float* Wv = (const float*)d_in[6];
  const float* bv = (const float*)d_in[7];
  const float* Wo = (const float*)d_in[8];
  const float* bo = (const float*)d_in[9];
  float* out = (float*)d_out;

  const size_t elems = (size_t)BATCH * NHEADS * SEQ * HEADD;  // 3,145,728 halves
  half_t* qws = (half_t*)d_ws;
  half_t* kws = qws + elems;
  half_t* vws = kws + elems;
  half_t* ctx = vws + elems;   // [B,S,H] f16 ; total ws use = 24 MB

  qkv_proj_kernel<<<dim3(64, 9), 256, 0, stream>>>(hs, Wq, bq, Wk, bk, Wv, bv,
                                                   qws, kws, vws);
  attn_kernel<<<dim3(32, NHEADS, BATCH), 128, 0, stream>>>(qws, kws, vws, mask, ctx);
  out_proj_kernel<<<dim3(64, 3), 256, 0, stream>>>(ctx, Wo, bo, out);
}
